// QwenAttention_83391085019786
// MI455X (gfx1250) — matrix operand order repaired and checked by reading, not compiled, not run on hardware
//
#include <hip/hip_runtime.h>
#include <hip/hip_bf16.h>
#include <math.h>

typedef __bf16 bf16_t;
typedef __attribute__((ext_vector_type(16))) __bf16 v16bf;
typedef __attribute__((ext_vector_type(8)))  __bf16 v8bf;
typedef __attribute__((ext_vector_type(8)))  float  v8f;

static constexpr int BATCH  = 2;
static constexpr int SEQ    = 2048;
static constexpr int NHEAD  = 16;
static constexpr int DHEAD  = 128;
static constexpr int NSTATE = 2048;
static constexpr int QKVN   = 3 * NHEAD * DHEAD; // 6144

// ---------------------------------------------------------------------------
// CDNA5 async memory->LDS DMA (ISA 15.18.3 op 98, ASYNCcnt-tracked)
// ---------------------------------------------------------------------------
static __device__ __forceinline__ unsigned lds_addr_of(void* p) {
  return (unsigned)(unsigned long long)(__attribute__((address_space(3))) char*)p;
}
static __device__ __forceinline__ void async_load_b128(void* lds_dst, const void* gsrc) {
  asm volatile("global_load_async_to_lds_b128 %0, %1, off"
               :: "v"(lds_addr_of(lds_dst)), "v"((unsigned long long)gsrc)
               : "memory");
}
static __device__ __forceinline__ void wait_async() {
  asm volatile("s_wait_asynccnt 0" ::: "memory");
}

// ---------------------------------------------------------------------------
// WMMA helpers (CDNA5 wave32, 16x16x32 bf16 -> f32)
// ---------------------------------------------------------------------------
static __device__ __forceinline__ v8f wmma_bf16(v16bf a, v16bf b, v8f c) {
  return __builtin_amdgcn_wmma_f32_16x16x32_bf16(
      /*neg_a=*/false, a, /*neg_b=*/false, b,
      /*c_mod=*/(short)0, c, /*reuse_a=*/false, /*reuse_b=*/false);
}

// A-matrix fragment: 16x32 bf16, row-major source with leading dim ld.
// ISA layout: lane<16 holds row=lane, K={0..7,16..23}; lane>=16 K={8..15,24..31}.
static __device__ __forceinline__ v16bf load_a_frag(const bf16_t* __restrict__ src, int ld) {
  const int lane = threadIdx.x & 31;
  const int row  = lane & 15;
  const int kh   = (lane >> 4) & 1;
  const bf16_t* r = src + (size_t)row * ld;
  v16bf f;
#pragma unroll
  for (int i = 0; i < 8; ++i) f[i] = r[kh * 8 + i];
#pragma unroll
  for (int i = 0; i < 8; ++i) f[8 + i] = r[16 + kh * 8 + i];
  return f;
}

// B-matrix fragment from B^T storage: src is [N][K] row-major (contiguous K).
// ISA layout: lane holds col n=lane&15; lane<16 K={0..7,16..23}; lane>=16 K={8..15,24..31}.
static __device__ __forceinline__ v16bf load_bt_frag(const bf16_t* __restrict__ src, int ld) {
  const int lane = threadIdx.x & 31;
  const bf16_t* r = src + (size_t)(lane & 15) * ld + ((lane >> 4) & 1) * 8;
  v16bf f;
#pragma unroll
  for (int i = 0; i < 16; ++i) f[i] = r[i < 8 ? i : 8 + i];
  return f;
}

// ---------------------------------------------------------------------------
// fp32 -> bf16 elementwise, 8 elements per thread (2x b128 in, 1x b128 out)
// ---------------------------------------------------------------------------
__global__ void f32_to_bf16_kernel(const float* __restrict__ in, bf16_t* __restrict__ out, long n) {
  long i = ((long)blockIdx.x * blockDim.x + threadIdx.x) * 8;
  const long stride = (long)gridDim.x * blockDim.x * 8;
  for (; i < n; i += stride) {
    const float4 a = *reinterpret_cast<const float4*>(in + i);
    const float4 b = *reinterpret_cast<const float4*>(in + i + 4);
    v8bf o;
    o[0] = (bf16_t)a.x; o[1] = (bf16_t)a.y; o[2] = (bf16_t)a.z; o[3] = (bf16_t)a.w;
    o[4] = (bf16_t)b.x; o[5] = (bf16_t)b.y; o[6] = (bf16_t)b.z; o[7] = (bf16_t)b.w;
    *reinterpret_cast<v8bf*>(out + i) = o;
  }
}

// ---------------------------------------------------------------------------
// GEMM: C[M,N] = A[M,K] * Bt[N,K]^T (+bias). 256 threads, 128x64 block tile,
// 8 waves (4M x 2N) each own a 32x32 tile (4 WMMA / K-step). Double-buffered
// LDS staged with async global->LDS DMA; one barrier per K-step.
// ---------------------------------------------------------------------------
__global__ __launch_bounds__(256) void gemm_bf16_kernel(
    const bf16_t* __restrict__ A, const bf16_t* __restrict__ Bt,
    const float* __restrict__ bias, float* __restrict__ Cf, bf16_t* __restrict__ Cb,
    int M, int N, int K) {
  __shared__ __align__(16) bf16_t As[2][128 * 32];
  __shared__ __align__(16) bf16_t Bs[2][64 * 32];
  const int tid  = threadIdx.x;
  const int lane = tid & 31;
  const int w    = tid >> 5;
  const int wm = w >> 1, wn = w & 1;
  const int m0 = blockIdx.y * 128, n0 = blockIdx.x * 64;

  // Staging map: chunks of 8 bf16 (16B). A: 512 chunks (2/thread), B: 256 (1/thread).
  const int ar0 = tid >> 2, ac = (tid & 3) * 8;     // A chunk 0: rows 0..63
  const int ar1 = ar0 + 64;                         // A chunk 1: rows 64..127
  const int br  = tid >> 2;                         // B rows 0..63 (tid<256)

  auto stage = [&](int buf, int kt) {
    async_load_b128(&As[buf][ar0 * 32 + ac], A + (size_t)(m0 + ar0) * K + kt + ac);
    async_load_b128(&As[buf][ar1 * 32 + ac], A + (size_t)(m0 + ar1) * K + kt + ac);
    async_load_b128(&Bs[buf][br * 32 + ac],  Bt + (size_t)(n0 + br) * K + kt + ac);
  };

  v8f acc00 = {}, acc01 = {}, acc10 = {}, acc11 = {};
  stage(0, 0);
  wait_async();
  __syncthreads();

  for (int kt = 0; kt < K; kt += 32) {
    const int cur = (kt >> 5) & 1;
    if (kt + 32 < K) stage(cur ^ 1, kt + 32);
    v16bf a0 = load_a_frag(As[cur] + (wm * 32) * 32, 32);
    v16bf a1 = load_a_frag(As[cur] + (wm * 32 + 16) * 32, 32);
    v16bf b0 = load_bt_frag(Bs[cur] + (wn * 32) * 32, 32);
    v16bf b1 = load_bt_frag(Bs[cur] + (wn * 32 + 16) * 32, 32);
    acc00 = wmma_bf16(a0, b0, acc00);
    acc01 = wmma_bf16(a0, b1, acc01);
    acc10 = wmma_bf16(a1, b0, acc10);
    acc11 = wmma_bf16(a1, b1, acc11);
    wait_async();
    __syncthreads();
  }

  const int half = (lane >> 4) & 1;
  const int ncol = n0 + wn * 32 + (lane & 15);
  v8f accs[2][2] = {{acc00, acc01}, {acc10, acc11}};
#pragma unroll
  for (int am = 0; am < 2; ++am) {
    const int mrow = m0 + wm * 32 + am * 16 + half * 8;  // C layout: row = vgpr + 8*half
#pragma unroll
    for (int r = 0; r < 8; ++r) {
      float v0 = accs[am][0][r], v1 = accs[am][1][r];
      const int m = mrow + r;
      if (bias) { v0 += bias[ncol]; v1 += bias[ncol + 16]; }
      if (Cf) { Cf[(size_t)m * N + ncol] = v0; Cf[(size_t)m * N + ncol + 16] = v1; }
      if (Cb) { Cb[(size_t)m * N + ncol] = (bf16_t)v0; Cb[(size_t)m * N + ncol + 16] = (bf16_t)v1; }
    }
  }
}

// ---------------------------------------------------------------------------
// RoPE + split: reads bf16 qkv [B,S,6144]; writes q/k/v bf16 in [B,H,S,D]
// plus fp32 k,v outputs in [B,S,H,D] (reference return layout).
// ---------------------------------------------------------------------------
__global__ __launch_bounds__(256) void rope_split_kernel(
    const bf16_t* __restrict__ qkv, const float* __restrict__ fr, const float* __restrict__ fi,
    bf16_t* __restrict__ Qo, bf16_t* __restrict__ Ko, bf16_t* __restrict__ Vo,
    float* __restrict__ KoutF, float* __restrict__ VoutF) {
  const int s = blockIdx.x, b = blockIdx.y;
  const size_t rowbase = (size_t)(b * SEQ + s) * QKVN;
  const size_t fbase   = (size_t)(b * SEQ + s) * (DHEAD / 2);
  for (int p = threadIdx.x; p < NHEAD * (DHEAD / 2); p += 256) {
    const int h = p >> 6, d = p & 63;
    const float cr = fr[fbase + d], ci = fi[fbase + d];
    const size_t qi0 = rowbase + h * DHEAD + d;
    const float qr = (float)qkv[qi0],              qim = (float)qkv[qi0 + 64];
    const float kr = (float)qkv[qi0 + NSTATE],     kim = (float)qkv[qi0 + NSTATE + 64];
    const float vr = (float)qkv[qi0 + 2 * NSTATE], vim = (float)qkv[qi0 + 2 * NSTATE + 64];
    const float qor = qr * cr - qim * ci, qoi = qr * ci + qim * cr;
    const float kor = kr * cr - kim * ci, koi = kr * ci + kim * cr;
    const size_t hb = ((size_t)(b * NHEAD + h) * SEQ + s) * DHEAD + d;
    Qo[hb] = (bf16_t)qor; Qo[hb + 64] = (bf16_t)qoi;
    Ko[hb] = (bf16_t)kor; Ko[hb + 64] = (bf16_t)koi;
    Vo[hb] = (bf16_t)vr;  Vo[hb + 64] = (bf16_t)vim;
    const size_t ob = ((size_t)(b * SEQ + s) * NHEAD + h) * DHEAD + d;
    KoutF[ob] = kor; KoutF[ob + 64] = koi;
    VoutF[ob] = vr;  VoutF[ob + 64] = vim;
  }
}

// ---------------------------------------------------------------------------
// Flash attention: grid (S/128, H, B); 8 waves x 16 q-rows; 32-key tiles.
// K staged via async DMA; V staged transposed (Vt[d][k]) so PV B-fragments
// are contiguous ds_load_b128 pairs instead of 16 scalar loads.
// ---------------------------------------------------------------------------
__global__ __launch_bounds__(256) void flash_attn_kernel(
    const bf16_t* __restrict__ Q, const bf16_t* __restrict__ Kc,
    const bf16_t* __restrict__ Vc, bf16_t* __restrict__ O) {
  __shared__ __align__(16) bf16_t Ks[32 * 128];   // [key][d]
  __shared__ __align__(16) bf16_t Vt[128 * 32];   // [d][key]  (transposed)
  __shared__ __align__(16) bf16_t Ps[8][16 * 32];
  const int tid = threadIdx.x, lane = tid & 31, w = tid >> 5;
  const int q0 = blockIdx.x * 128, h = blockIdx.y, b = blockIdx.z;
  const int qw = q0 + w * 16;
  const size_t headbase = (size_t)(b * NHEAD + h) * SEQ * DHEAD;
  const int half = (lane >> 4) & 1;
  const int qrow_base = qw + half * 8;
  const int kcol_lane = lane & 15;
  const float scale = 0.08838834764831845f;  // 1/sqrt(128)

  v16bf qf[4];
#pragma unroll
  for (int kk = 0; kk < 4; ++kk)
    qf[kk] = load_a_frag(Q + headbase + (size_t)qw * DHEAD + kk * 32, DHEAD);

  v8f acc[8] = {};
  float mst[8], lst[8];
#pragma unroll
  for (int r = 0; r < 8; ++r) { mst[r] = -1e30f; lst[r] = 0.f; }

  const int kmax = q0 + 128;  // causal: last key tile is the block's own rows
  for (int kb = 0; kb < kmax; kb += 32) {
    // K tile: async DMA, 2 chunks of 8 bf16 per thread.
#pragma unroll
    for (int c = tid; c < 512; c += 256) {
      const int row = c >> 4, doff = (c & 15) * 8;
      async_load_b128(&Ks[row * 128 + doff],
                      Kc + headbase + (size_t)(kb + row) * DHEAD + doff);
    }
    // V tile: load b128 and scatter transposed into Vt.
    for (int c = tid; c < 512; c += 256) {
      const int row = c >> 4, doff = (c & 15) * 8;
      const uint4 u = *reinterpret_cast<const uint4*>(
          Vc + headbase + (size_t)(kb + row) * DHEAD + doff);
      const bf16_t* vv = reinterpret_cast<const bf16_t*>(&u);
#pragma unroll
      for (int j = 0; j < 8; ++j) Vt[(doff + j) * 32 + row] = vv[j];
    }
    wait_async();
    __syncthreads();

    v8f s0 = {}, s1 = {};
#pragma unroll
    for (int kk = 0; kk < 4; ++kk) {  // S = Q * K^T over D=128
      v16bf k0 = load_bt_frag(Ks + kk * 32, 128);
      v16bf k1 = load_bt_frag(Ks + 16 * 128 + kk * 32, 128);
      s0 = wmma_bf16(qf[kk], k0, s0);
      s1 = wmma_bf16(qf[kk], k1, s1);
    }

    float rowm[8];
#pragma unroll
    for (int r = 0; r < 8; ++r) {
      const int qr = qrow_base + r;
      float v0 = s0[r] * scale, v1 = s1[r] * scale;
      if (kb + kcol_lane > qr)      v0 = -1e30f;
      if (kb + 16 + kcol_lane > qr) v1 = -1e30f;
      s0[r] = v0; s1[r] = v1;
      rowm[r] = fmaxf(v0, v1);
    }
#pragma unroll
    for (int r = 0; r < 8; ++r)
#pragma unroll
      for (int mk = 1; mk < 16; mk <<= 1)
        rowm[r] = fmaxf(rowm[r], __shfl_xor(rowm[r], mk, 32));

    float corr[8], rs[8];
#pragma unroll
    for (int r = 0; r < 8; ++r) {
      const float mn = fmaxf(mst[r], rowm[r]);
      corr[r] = __expf(mst[r] - mn);
      mst[r] = mn;
      s0[r] = __expf(s0[r] - mn);
      s1[r] = __expf(s1[r] - mn);
      rs[r] = s0[r] + s1[r];
    }
#pragma unroll
    for (int r = 0; r < 8; ++r) {
#pragma unroll
      for (int mk = 1; mk < 16; mk <<= 1) rs[r] += __shfl_xor(rs[r], mk, 32);
      lst[r] = lst[r] * corr[r] + rs[r];
    }
#pragma unroll
    for (int nt = 0; nt < 8; ++nt)
#pragma unroll
      for (int r = 0; r < 8; ++r) acc[nt][r] *= corr[r];

    // Re-lay-out P (C layout -> A layout) through per-wave LDS scratch.
    bf16_t* pp = Ps[w];
#pragma unroll
    for (int r = 0; r < 8; ++r) {
      const int prow = r + half * 8;
      pp[prow * 32 + kcol_lane]      = (bf16_t)s0[r];
      pp[prow * 32 + 16 + kcol_lane] = (bf16_t)s1[r];
    }
    v16bf pa = load_a_frag(pp, 32);
#pragma unroll
    for (int nt = 0; nt < 8; ++nt) {  // O += P * V, 8 tiles across D
      v16bf vf = load_bt_frag(Vt + (nt * 16) * 32, 32);
      acc[nt] = wmma_bf16(pa, vf, acc[nt]);
    }
    __syncthreads();
  }

#pragma unroll
  for (int nt = 0; nt < 8; ++nt)
#pragma unroll
    for (int r = 0; r < 8; ++r) {
      const int q = qrow_base + r;
      const int dd = nt * 16 + kcol_lane;
      const float o = acc[nt][r] / lst[r];
      O[(size_t)(b * SEQ + q) * NSTATE + h * DHEAD + dd] = (bf16_t)o;
    }
}

// ---------------------------------------------------------------------------
extern "C" void kernel_launch(void* const* d_in, const int* in_sizes, int n_in,
                              void* d_out, int out_size, void* d_ws, size_t ws_size,
                              hipStream_t stream) {
  (void)in_sizes; (void)n_in; (void)out_size; (void)ws_size;
  const float* x     = (const float*)d_in[0];
  const float* f_r   = (const float*)d_in[1];
  const float* f_i   = (const float*)d_in[2];
  // d_in[3] = attention_mask: unused, causal mask is analytic
  const float* W_qkv = (const float*)d_in[4];
  const float* b_qkv = (const float*)d_in[5];
  const float* W_o   = (const float*)d_in[6];

  float* out   = (float*)d_out;
  float* koutF = out + (size_t)BATCH * SEQ * NSTATE;
  float* voutF = koutF + (size_t)BATCH * SEQ * NHEAD * DHEAD;

  char* ws = (char*)d_ws;
  size_t off = 0;
  auto alloc = [&](size_t bytes) -> void* {
    void* p = ws + off;
    off = (off + bytes + 255) & ~(size_t)255;
    return p;
  };
  bf16_t* xb    = (bf16_t*)alloc((size_t)BATCH * SEQ * NSTATE * 2);
  bf16_t* wqkvb = (bf16_t*)alloc((size_t)QKVN * NSTATE * 2);
  bf16_t* wob   = (bf16_t*)alloc((size_t)NSTATE * NSTATE * 2);
  bf16_t* qkvb  = (bf16_t*)alloc((size_t)BATCH * SEQ * QKVN * 2);
  bf16_t* qb    = (bf16_t*)alloc((size_t)BATCH * NHEAD * SEQ * DHEAD * 2);
  bf16_t* kb    = (bf16_t*)alloc((size_t)BATCH * NHEAD * SEQ * DHEAD * 2);
  bf16_t* vb    = (bf16_t*)alloc((size_t)BATCH * NHEAD * SEQ * DHEAD * 2);
  bf16_t* attnb = (bf16_t*)alloc((size_t)BATCH * SEQ * NSTATE * 2);

  f32_to_bf16_kernel<<<1024, 256, 0, stream>>>(x, xb, (long)BATCH * SEQ * NSTATE);
  f32_to_bf16_kernel<<<1024, 256, 0, stream>>>(W_qkv, wqkvb, (long)QKVN * NSTATE);
  f32_to_bf16_kernel<<<1024, 256, 0, stream>>>(W_o, wob, (long)NSTATE * NSTATE);

  gemm_bf16_kernel<<<dim3(QKVN / 64, (BATCH * SEQ) / 128), 256, 0, stream>>>(
      xb, wqkvb, b_qkv, nullptr, qkvb, BATCH * SEQ, QKVN, NSTATE);

  rope_split_kernel<<<dim3(SEQ, BATCH), 256, 0, stream>>>(
      qkvb, f_r, f_i, qb, kb, vb, koutF, voutF);

  flash_attn_kernel<<<dim3(SEQ / 128, NHEAD, BATCH), 256, 0, stream>>>(qb, kb, vb, attnb);

  gemm_bf16_kernel<<<dim3(NSTATE / 64, (BATCH * SEQ) / 128), 256, 0, stream>>>(
      attnb, wob, nullptr, out, nullptr, BATCH * SEQ, NSTATE, NSTATE);
}
